// ScopeGrammarLanguageModel_48773648613660
// MI455X (gfx1250) — compile-verified, matrix operand order checked by reading, not run on hardware
//
#include <hip/hip_runtime.h>
#include <hip/hip_bf16.h>
#include <math.h>

// Problem dims (from reference): B=32, T=512, H=256, E=256, S=50, V=10000, L=2
constexpr int kB = 32;
constexpr int kT = 512;
constexpr int kH = 256;
constexpr int kE = 256;
constexpr int kS = 50;
constexpr int kV = 10000;

typedef __bf16 bf16_t;
typedef __attribute__((ext_vector_type(16))) __bf16 v16bf;
typedef __attribute__((ext_vector_type(8)))  float  v8f;

__device__ __forceinline__ float sigf(float x) { return 1.0f / (1.0f + expf(-x)); }

// ---------------------------------------------------------------------------
// WMMA helpers (wave32, V_WMMA_F32_16X16X32_BF16)
// A tile (16x32, bf16): lanes 0-15 hold rows r0..r0+15 with K = k0+{0..7,16..23},
// lanes 16-31 hold same rows with K = k0+{8..15,24..31}. B is loaded from a
// row-major W[N,K] (we compute X * W^T), mirrored layout with N in lanes.
// C/D tile (16x16, f32): 8 VGPRs; lanes 0-15: N=lane, M=r0+{0..7};
// lanes 16-31: N=lane-16, M=r0+{8..15}.
// ---------------------------------------------------------------------------
__device__ __forceinline__ v16bf load_tile16(const bf16_t* __restrict__ base,
                                             int ld, int r0, int k0) {
  const int lane = threadIdx.x & 31;
  const int r  = r0 + (lane & 15);
  const int kb = k0 + ((lane >> 4) << 3);
  const bf16_t* q = base + r * ld + kb;
  v16bf v;
#pragma unroll
  for (int i = 0; i < 8; ++i) { v[i] = q[i]; v[i + 8] = q[i + 16]; }
  return v;
}

__device__ __forceinline__ v8f wmma_bf16(v16bf a, v16bf b, v8f c) {
  return __builtin_amdgcn_wmma_f32_16x16x32_bf16(false, a, false, b, (short)0, c,
                                                 false, false);
}

// ---------------------------------------------------------------------------
// Utility kernels
// ---------------------------------------------------------------------------
__global__ void k_cvt(const float* __restrict__ s, bf16_t* __restrict__ d, int n) {
  int i = blockIdx.x * blockDim.x + threadIdx.x;
  if (i < n) d[i] = (bf16_t)s[i];
}

__global__ void k_init_state(const float* __restrict__ h0, const float* __restrict__ c0,
                             float* __restrict__ h, float* __restrict__ c,
                             bf16_t* __restrict__ hbf, int n) {
  int i = blockIdx.x * blockDim.x + threadIdx.x;
  if (i < n) { float hv = h0[i]; h[i] = hv; hbf[i] = (bf16_t)hv; c[i] = c0[i]; }
}

__global__ void k_zero(float* __restrict__ p, int n) {
  int i = blockIdx.x * blockDim.x + threadIdx.x;
  if (i < n) p[i] = 0.f;
}

// ---------------------------------------------------------------------------
// Per-step kernel 1: scope gather + scope transform + identifier select.
// One workgroup (512 threads = 16 waves). Produces xin (bf16, [B,E]).
//   xcat = [stack[b,sidx[b]], emb[tok[b]]]      (B x 512)
//   z    = relu(xcat @ st_w1^T + st_b1)         (B x 256)
//   xin  = is_id ? (z @ st_w2^T + st_b2) : emb  (B x 256)
// ---------------------------------------------------------------------------
__global__ __launch_bounds__(512) void k_scope(
    const int* __restrict__ tokens, const int* __restrict__ scope_idx,
    const int* __restrict__ is_id,
    const float* __restrict__ emb_table,
    const bf16_t* __restrict__ wst1, const float* __restrict__ st_b1,
    const bf16_t* __restrict__ wst2, const float* __restrict__ st_b2,
    const float* __restrict__ stack_cur,
    bf16_t* __restrict__ xin_bf, float* __restrict__ emb_f32, int t) {
  __shared__ bf16_t xcat[kB][kH + kE];  // 32 KB
  __shared__ bf16_t zsh[kB][kH];        // 16 KB
  const int tid = threadIdx.x;
  for (int idx = tid; idx < kB * 512; idx += 512) {
    int b = idx >> 9, k = idx & 511;
    float v;
    if (k < kH) {
      int s = scope_idx[b * kT + t];
      v = stack_cur[(b * kS + s) * kH + k];
    } else {
      int tok = tokens[b * kT + t];
      v = emb_table[tok * kE + (k - kH)];
      emb_f32[b * kE + (k - kH)] = v;
    }
    xcat[b][k] = (bf16_t)v;
  }
  __syncthreads();
  const int wave = tid >> 5, lane = tid & 31;
  const int nl = lane & 15, mh = (lane >> 4) << 3;
  // z = relu(xcat @ wst1^T + b1):  M=32, N=256, K=512
  for (int job = wave; job < 32; job += 16) {
    const int mt = job & 1, nt = job >> 1;
    v8f acc = {};
    for (int k0 = 0; k0 < 512; k0 += 32) {
      v16bf a = load_tile16(&xcat[0][0], 512, mt * 16, k0);
      v16bf w = load_tile16(wst1, 512, nt * 16, k0);
      acc = wmma_bf16(a, w, acc);
    }
    const int n = nt * 16 + nl;
    const float bias = st_b1[n];
#pragma unroll
    for (int r = 0; r < 8; ++r) {
      float z = acc[r] + bias;
      zsh[mt * 16 + mh + r][n] = (bf16_t)(z > 0.f ? z : 0.f);
    }
  }
  __syncthreads();
  // scoped = z @ wst2^T + b2; xin = select
  for (int job = wave; job < 32; job += 16) {
    const int mt = job & 1, nt = job >> 1;
    v8f acc = {};
    for (int k0 = 0; k0 < kH; k0 += 32) {
      v16bf a = load_tile16(&zsh[0][0], kH, mt * 16, k0);
      v16bf w = load_tile16(wst2, kH, nt * 16, k0);
      acc = wmma_bf16(a, w, acc);
    }
    const int n = nt * 16 + nl;
    const float bias = st_b2[n];
#pragma unroll
    for (int r = 0; r < 8; ++r) {
      int b = mt * 16 + mh + r;
      float scoped = acc[r] + bias;
      float x = (is_id[b * kT + t] != 0) ? scoped : emb_f32[b * kE + n];
      xin_bf[b * kE + n] = (bf16_t)x;
    }
  }
}

// ---------------------------------------------------------------------------
// Per-step kernel 2: both LSTM layers fused. One workgroup (512 threads).
// gates = [x|h] @ [W_ih|W_hh]^T; i/f/g/o accumulated by the same wave so the
// cell nonlinearity is applied locally. State (h,c) frozen for inactive rows;
// unmasked h0 feeds layer 1; unmasked h1 is the step output (feats).
// ---------------------------------------------------------------------------
__global__ __launch_bounds__(512) void k_lstm(
    const bf16_t* __restrict__ xin_bf,
    const bf16_t* __restrict__ wih0, const bf16_t* __restrict__ whh0,
    const float* __restrict__ bih0, const float* __restrict__ bhh0,
    const bf16_t* __restrict__ wih1, const bf16_t* __restrict__ whh1,
    const float* __restrict__ bih1, const float* __restrict__ bhh1,
    float* __restrict__ hstate, float* __restrict__ cstate,
    bf16_t* __restrict__ h_bf,
    float* __restrict__ h1u_f32, bf16_t* __restrict__ feats_bf,
    const int* __restrict__ lengths, int t) {
  __shared__ bf16_t xh[kB][2 * kH];  // 32 KB: [x | h_layer]
  __shared__ bf16_t hun[kB][kH];     // 16 KB: unmasked h of current layer
  const int tid = threadIdx.x, wave = tid >> 5, lane = tid & 31;
  const int nl = lane & 15, mh = (lane >> 4) << 3;
  for (int idx = tid; idx < kB * 512; idx += 512) {
    int b = idx >> 9, k = idx & 511;
    xh[b][k] = (k < kH) ? xin_bf[b * kH + k] : h_bf[b * kH + (k - kH)];
  }
  __syncthreads();
  for (int layer = 0; layer < 2; ++layer) {
    if (layer == 1) {
      __syncthreads();
      for (int idx = tid; idx < kB * 512; idx += 512) {
        int b = idx >> 9, k = idx & 511;
        xh[b][k] = (k < kH) ? hun[b][k] : h_bf[(kB + b) * kH + (k - kH)];
      }
      __syncthreads();
    }
    const bf16_t* wih = layer ? wih1 : wih0;
    const bf16_t* whh = layer ? whh1 : whh0;
    const float* bi = layer ? bih1 : bih0;
    const float* bh = layer ? bhh1 : bhh0;
    for (int job = wave; job < 32; job += 16) {
      const int mt = job & 1, nt = job >> 1;
      v8f ai = {}, af = {}, ag = {}, ao = {};
      for (int k0 = 0; k0 < 512; k0 += 32) {
        v16bf a = load_tile16(&xh[0][0], 512, mt * 16, k0);
        const bf16_t* w = (k0 < kH) ? wih : whh;
        const int kk = (k0 < kH) ? k0 : k0 - kH;
        v16bf wi = load_tile16(w, kH, 0 * kH + nt * 16, kk);
        v16bf wf = load_tile16(w, kH, 1 * kH + nt * 16, kk);
        v16bf wg = load_tile16(w, kH, 2 * kH + nt * 16, kk);
        v16bf wo = load_tile16(w, kH, 3 * kH + nt * 16, kk);
        ai = wmma_bf16(a, wi, ai);
        af = wmma_bf16(a, wf, af);
        ag = wmma_bf16(a, wg, ag);
        ao = wmma_bf16(a, wo, ao);
      }
      const int n = nt * 16 + nl;
      const float b_i = bi[n] + bh[n];
      const float b_f = bi[kH + n] + bh[kH + n];
      const float b_g = bi[2 * kH + n] + bh[2 * kH + n];
      const float b_o = bi[3 * kH + n] + bh[3 * kH + n];
#pragma unroll
      for (int r = 0; r < 8; ++r) {
        const int b = mt * 16 + mh + r;
        const int si = (layer * kB + b) * kH + n;
        float iv = sigf(ai[r] + b_i);
        float fv = sigf(af[r] + b_f);
        float gv = tanhf(ag[r] + b_g);
        float ov = sigf(ao[r] + b_o);
        float cold = cstate[si];
        float hold = hstate[si];
        float cn = fv * cold + iv * gv;
        float hn = ov * tanhf(cn);
        bool act = t < lengths[b];
        float cm = act ? cn : cold;
        float hm = act ? hn : hold;
        cstate[si] = cm;
        hstate[si] = hm;
        h_bf[si] = (bf16_t)hm;
        hun[b][n] = (bf16_t)hn;  // unmasked feeds next layer
        if (layer == 1) {
          h1u_f32[b * kH + n] = hn;
          feats_bf[(b * kT + t) * kH + n] = (bf16_t)hn;  // out stored unmasked
        }
      }
    }
  }
}

// ---------------------------------------------------------------------------
// Per-step kernel 3: stack gate+value update. 100 blocks x 256 threads; each
// block owns 16 rows of the flattened (b*S+s) stack. Writes to ping-pong
// buffer stack_next (no cross-block RAW hazard).
// ---------------------------------------------------------------------------
__global__ __launch_bounds__(256) void k_stack(
    const float* __restrict__ stack_cur, float* __restrict__ stack_next,
    const bf16_t* __restrict__ wval, const float* __restrict__ val_b,
    const float* __restrict__ gate_w, const float* __restrict__ gate_b,
    const float* __restrict__ h1u_f32,
    const int* __restrict__ update_mask, const int* __restrict__ lengths, int t) {
  __shared__ bf16_t arow[16][512];  // 16 KB: si rows in bf16
  __shared__ float gred[16][16];
  __shared__ float gsh[16];
  const int tid = threadIdx.x;
  const int m0 = blockIdx.x * 16;
  {  // stage si rows + per-row gate dot partials (16 threads per row)
    const int r = tid >> 4, j = tid & 15;
    const int m = m0 + r;
    const int b = m / kS;
    float acc = 0.f;
#pragma unroll
    for (int i = 0; i < 32; ++i) {
      int k = j * 32 + i;
      float v = (k < kH) ? stack_cur[m * kH + k] : h1u_f32[b * kH + (k - kH)];
      arow[r][k] = (bf16_t)v;
      acc += gate_w[k] * v;
    }
    gred[r][j] = acc;
  }
  __syncthreads();
  if (tid < 16) {
    float s = gate_b[0];
#pragma unroll
    for (int j = 0; j < 16; ++j) s += gred[tid][j];
    gsh[tid] = 1.0f / (1.0f + expf(-s));
  }
  __syncthreads();
  const int wave = tid >> 5, lane = tid & 31;
  const int nl = lane & 15, mh = (lane >> 4) << 3;
  for (int nt = wave; nt < 16; nt += 8) {
    v8f acc = {};
    for (int k0 = 0; k0 < 512; k0 += 32) {
      v16bf a = load_tile16(&arow[0][0], 512, 0, k0);
      v16bf w = load_tile16(wval, 512, nt * 16, k0);
      acc = wmma_bf16(a, w, acc);
    }
    const int n = nt * 16 + nl;
    const float bias = val_b[n];
#pragma unroll
    for (int r = 0; r < 8; ++r) {
      const int m = m0 + mh + r;
      const int b = m / kS, s = m % kS;
      float val = tanhf(acc[r] + bias);
      float g = gsh[mh + r];
      float st = stack_cur[m * kH + n];
      float ns = st * (1.0f - g) + g * val;
      if (update_mask[(b * kT + t) * kS + s] > 0) ns = 0.f;
      bool act = t < lengths[b];
      stack_next[m * kH + n] = act ? ns : st;
    }
  }
}

// ---------------------------------------------------------------------------
// Output head: g1 = relu(feats @ out_w1^T + b1); logits = g1 @ out_w2^T + b2
// masked with -1.0 for padded positions. HBM-bound (655 MB store).
// ---------------------------------------------------------------------------
__global__ __launch_bounds__(256) void k_head1(
    const bf16_t* __restrict__ feats_bf, const bf16_t* __restrict__ wout1,
    const float* __restrict__ out_b1, bf16_t* __restrict__ g1_bf) {
  const int tid = threadIdx.x, wave = tid >> 5, lane = tid & 31;
  const int nl = lane & 15, mh = (lane >> 4) << 3;
  const int m0 = blockIdx.y * 16;
  const int nt = blockIdx.x * 8 + wave;
  v8f acc = {};
  for (int k0 = 0; k0 < kH; k0 += 32) {
    v16bf a = load_tile16(feats_bf, kH, m0, k0);
    v16bf w = load_tile16(wout1, kH, nt * 16, k0);
    acc = wmma_bf16(a, w, acc);
  }
  const int n = nt * 16 + nl;
  const float bias = out_b1[n];
#pragma unroll
  for (int r = 0; r < 8; ++r) {
    float z = acc[r] + bias;
    g1_bf[(m0 + mh + r) * kH + n] = (bf16_t)(z > 0.f ? z : 0.f);
  }
}

__global__ __launch_bounds__(256) void k_head2(
    const bf16_t* __restrict__ g1_bf, const bf16_t* __restrict__ wout2,
    const float* __restrict__ out_b2, const int* __restrict__ lengths,
    float* __restrict__ out) {
  const int tid = threadIdx.x, wave = tid >> 5, lane = tid & 31;
  const int nl = lane & 15, mh = (lane >> 4) << 3;
  const int m0 = blockIdx.y * 16;
  const int nt = blockIdx.x * 8 + wave;
  if (nt >= kV / 16) return;  // wave-uniform guard (EXEC stays full)
  v8f acc = {};
  for (int k0 = 0; k0 < kH; k0 += 32) {
    v16bf a = load_tile16(g1_bf, kH, m0, k0);
    v16bf w = load_tile16(wout2, kH, nt * 16, k0);
    acc = wmma_bf16(a, w, acc);
  }
  const int n = nt * 16 + nl;
  const float bias = out_b2[n];
#pragma unroll
  for (int r = 0; r < 8; ++r) {
    const int m = m0 + mh + r;
    const int b = m >> 9, ts = m & 511;  // m = b*T + t, T=512
    bool act = ts < lengths[b];
    out[(size_t)m * kV + n] = act ? (acc[r] + bias) : -1.0f;
  }
}

// ---------------------------------------------------------------------------
extern "C" void kernel_launch(void* const* d_in, const int* in_sizes, int n_in,
                              void* d_out, int out_size, void* d_ws, size_t ws_size,
                              hipStream_t stream) {
  (void)in_sizes; (void)n_in; (void)out_size; (void)ws_size;
  const int*   tokens  = (const int*)d_in[0];
  const int*   scope_i = (const int*)d_in[1];
  const int*   is_id   = (const int*)d_in[2];
  const int*   umask   = (const int*)d_in[3];
  const int*   lengths = (const int*)d_in[4];
  const float* emb     = (const float*)d_in[5];
  const float* w_ih0   = (const float*)d_in[6];
  const float* w_hh0   = (const float*)d_in[7];
  const float* b_ih0   = (const float*)d_in[8];
  const float* b_hh0   = (const float*)d_in[9];
  const float* w_ih1   = (const float*)d_in[10];
  const float* w_hh1   = (const float*)d_in[11];
  const float* b_ih1   = (const float*)d_in[12];
  const float* b_hh1   = (const float*)d_in[13];
  const float* st_w1   = (const float*)d_in[14];
  const float* st_b1   = (const float*)d_in[15];
  const float* st_w2   = (const float*)d_in[16];
  const float* st_b2   = (const float*)d_in[17];
  const float* gate_w  = (const float*)d_in[18];
  const float* gate_b  = (const float*)d_in[19];
  const float* val_w   = (const float*)d_in[20];
  const float* val_b   = (const float*)d_in[21];
  const float* out_w1  = (const float*)d_in[22];
  const float* out_b1  = (const float*)d_in[23];
  const float* out_w2  = (const float*)d_in[24];
  const float* out_b2  = (const float*)d_in[25];
  const float* h0      = (const float*)d_in[26];
  const float* c0      = (const float*)d_in[27];
  float* out = (float*)d_out;

  // ---- carve workspace (~28 MB) ----
  char* p = (char*)d_ws;
  auto take = [&](size_t bytes) { char* r = p; p += (bytes + 255) & ~(size_t)255; return r; };
  bf16_t* wst1_bf  = (bf16_t*)take(sizeof(bf16_t) * kH * (kE + kH));
  bf16_t* wst2_bf  = (bf16_t*)take(sizeof(bf16_t) * kE * kH);
  bf16_t* wih0_bf  = (bf16_t*)take(sizeof(bf16_t) * 4 * kH * kE);
  bf16_t* whh0_bf  = (bf16_t*)take(sizeof(bf16_t) * 4 * kH * kH);
  bf16_t* wih1_bf  = (bf16_t*)take(sizeof(bf16_t) * 4 * kH * kH);
  bf16_t* whh1_bf  = (bf16_t*)take(sizeof(bf16_t) * 4 * kH * kH);
  bf16_t* wval_bf  = (bf16_t*)take(sizeof(bf16_t) * kH * 2 * kH);
  bf16_t* wout1_bf = (bf16_t*)take(sizeof(bf16_t) * kH * kH);
  bf16_t* wout2_bf = (bf16_t*)take(sizeof(bf16_t) * kV * kH);
  bf16_t* feats_bf = (bf16_t*)take(sizeof(bf16_t) * kB * kT * kH);
  bf16_t* g1_bf    = (bf16_t*)take(sizeof(bf16_t) * kB * kT * kH);
  bf16_t* hbf      = (bf16_t*)take(sizeof(bf16_t) * 2 * kB * kH);
  bf16_t* xin_bf   = (bf16_t*)take(sizeof(bf16_t) * kB * kE);
  float* hstate = (float*)take(sizeof(float) * 2 * kB * kH);
  float* cstate = (float*)take(sizeof(float) * 2 * kB * kH);
  float* stackA = (float*)take(sizeof(float) * kB * kS * kH);
  float* stackB = (float*)take(sizeof(float) * kB * kS * kH);
  float* emb_f  = (float*)take(sizeof(float) * kB * kE);
  float* h1u_f  = (float*)take(sizeof(float) * kB * kH);

  // ---- one-time weight conversion to bf16 ----
  auto cvt = [&](const float* s, bf16_t* d, int n) {
    k_cvt<<<(n + 255) / 256, 256, 0, stream>>>(s, d, n);
  };
  cvt(st_w1,  wst1_bf,  kH * (kE + kH));
  cvt(st_w2,  wst2_bf,  kE * kH);
  cvt(w_ih0,  wih0_bf,  4 * kH * kE);
  cvt(w_hh0,  whh0_bf,  4 * kH * kH);
  cvt(w_ih1,  wih1_bf,  4 * kH * kH);
  cvt(w_hh1,  whh1_bf,  4 * kH * kH);
  cvt(val_w,  wval_bf,  kH * 2 * kH);
  cvt(out_w1, wout1_bf, kH * kH);
  cvt(out_w2, wout2_bf, kV * kH);

  // ---- init recurrent state (re-done every call: graph-replay safe) ----
  k_init_state<<<(2 * kB * kH + 255) / 256, 256, 0, stream>>>(
      h0, c0, hstate, cstate, hbf, 2 * kB * kH);
  k_zero<<<(kB * kS * kH + 255) / 256, 256, 0, stream>>>(stackA, kB * kS * kH);

  // ---- sequential scan over T timesteps ----
  for (int t = 0; t < kT; ++t) {
    const float* scur = (t & 1) ? stackB : stackA;
    float*       snxt = (t & 1) ? stackA : stackB;
    k_scope<<<1, 512, 0, stream>>>(tokens, scope_i, is_id, emb,
                                   wst1_bf, st_b1, wst2_bf, st_b2,
                                   scur, xin_bf, emb_f, t);
    k_lstm<<<1, 512, 0, stream>>>(xin_bf, wih0_bf, whh0_bf, b_ih0, b_hh0,
                                  wih1_bf, whh1_bf, b_ih1, b_hh1,
                                  hstate, cstate, hbf, h1u_f, feats_bf,
                                  lengths, t);
    k_stack<<<(kB * kS) / 16, 256, 0, stream>>>(scur, snxt, wval_bf, val_b,
                                                gate_w, gate_b, h1u_f,
                                                umask, lengths, t);
  }

  // ---- output head ----
  k_head1<<<dim3(kH / 128, (kB * kT) / 16), 256, 0, stream>>>(
      feats_bf, wout1_bf, out_b1, g1_bf);
  k_head2<<<dim3((kV / 16 + 7) / 8, (kB * kT) / 16), 256, 0, stream>>>(
      g1_bf, wout2_bf, out_b2, lengths, out);
}
